// GQAFractalAttentionAdvanced_30709016167057
// MI455X (gfx1250) — compile-verified, matrix operand order checked by reading, not run on hardware
//
#include <hip/hip_runtime.h>
#include <hip/hip_bf16.h>
#include <stdint.h>

typedef unsigned short u16;
typedef __attribute__((ext_vector_type(16))) __bf16 v16bf;
typedef __attribute__((ext_vector_type(8)))  __bf16 v8bf;
typedef __attribute__((ext_vector_type(8)))  float  v8f;
typedef __attribute__((ext_vector_type(4)))  unsigned int v4u;
typedef __attribute__((ext_vector_type(8)))  int v8i;
typedef __attribute__((ext_vector_type(4)))  int v4i;

#define DIM   1024
#define SEQ   2048
#define BATCH 2
#define QG    2
#define HPG   4
#define HD    128
#define GD    512            // per-group feature dim (HPG*HD)
#define NKEY  (QG*SEQ)       // concatenated key axis, G-major
#define SCALE 0.08838834764831845f   // 128^-0.5
#define LN_EPS 1e-5f

static __device__ __forceinline__ u16 f32_to_bf16(float f){
  union { float f; unsigned u; } v; v.f = f;
  unsigned r = v.u + 0x7FFFu + ((v.u >> 16) & 1u);   // round-to-nearest-even
  return (u16)(r >> 16);
}
static __device__ __forceinline__ __bf16 us_as_bf(u16 u){
  union { u16 u; __bf16 b; } c; c.u = u; return c.b;
}
// A/B fragment for v_wmma_f32_16x16x32_bf16. `p` already includes the
// +8-element offset for lanes 16..31; chunks are K{base..base+7} and
// K{base+16..base+23} -> two 16-byte loads.
static __device__ __forceinline__ v16bf ld_frag(const u16* p){
  v8bf lo = *(const v8bf*)p;
  v8bf hi = *(const v8bf*)(p + 16);
  v16bf f;
#pragma unroll
  for (int i = 0; i < 8; ++i){ f[i] = lo[i]; f[i+8] = hi[i]; }
  return f;
}
static __device__ __forceinline__ v8f wmma_bf16(v16bf a, v16bf b, v8f c){
  return __builtin_amdgcn_wmma_f32_16x16x32_bf16(false, a, false, b, (short)0, c, false, false);
}

// ---------------------------------------------------------------- conversions
__global__ void cvt_f32_bf16_kernel(const float* __restrict__ in,
                                    u16* __restrict__ out, int n){
  for (int i = blockIdx.x*blockDim.x + threadIdx.x; i < n; i += gridDim.x*blockDim.x)
    out[i] = f32_to_bf16(in[i]);
}

// ---------------------------------------------------------------- GEMM
// C = A[M,K] @ W[N,K]^T. 64x64 tile per 128-thread block; wave w owns a
// 16-row stripe (4 C tiles). Register double-buffered k-loop so the next
// chunk's fragment loads issue before this chunk's WMMAs (load/WMMA overlap).
// mode 0: store bf16 to outB[M,N]               (Q projection)
// mode 1: scatter K grouped-cols + V transposed (KV projection)
// mode 2: store f32 to outF[M,N]                (output projection)
__global__ void gemm_bf16_kernel(const u16* __restrict__ A, const u16* __restrict__ W,
                                 int M, int N, int K, int mode,
                                 u16* __restrict__ outB, u16* __restrict__ Kb,
                                 u16* __restrict__ Vt, float* __restrict__ outF){
  const int tid = threadIdx.x, wv = tid >> 5, ln = tid & 31;
  const int lane16 = ln & 15, hi8 = (ln >= 16) ? 8 : 0;
  const int m0 = blockIdx.x*64 + wv*16;
  const int n0 = blockIdx.y*64;
  const u16* arow = A + (size_t)(m0 + lane16)*K + hi8;
  const u16* wrow[4];
#pragma unroll
  for (int t = 0; t < 4; ++t) wrow[t] = W + (size_t)(n0 + t*16 + lane16)*K + hi8;
  v8f acc[4];
#pragma unroll
  for (int t = 0; t < 4; ++t)
#pragma unroll
    for (int v = 0; v < 8; ++v) acc[t][v] = 0.f;

  // prime the pipeline
  v16bf af = ld_frag(arow);
  v16bf bf[4];
#pragma unroll
  for (int t = 0; t < 4; ++t) bf[t] = ld_frag(wrow[t]);

  for (int k = 32; k <= K; k += 32){
    v16bf an = af;
    v16bf bn[4];
#pragma unroll
    for (int t = 0; t < 4; ++t) bn[t] = bf[t];
    if (k < K){                       // issue next-chunk loads first
      an = ld_frag(arow + k);
#pragma unroll
      for (int t = 0; t < 4; ++t) bn[t] = ld_frag(wrow[t] + k);
    }
#pragma unroll
    for (int t = 0; t < 4; ++t) acc[t] = wmma_bf16(af, bf[t], acc[t]);
    af = an;
#pragma unroll
    for (int t = 0; t < 4; ++t) bf[t] = bn[t];
  }

#pragma unroll
  for (int t = 0; t < 4; ++t){
    const int col = n0 + t*16 + lane16;
#pragma unroll
    for (int v = 0; v < 8; ++v){
      const int row = m0 + v + hi8;              // C layout: M = v (+8 for hi lanes)
      const float val = acc[t][v];
      if (mode == 0){
        outB[(size_t)row*N + col] = f32_to_bf16(val);
      } else if (mode == 1){
        const int b = row >> 11, lk = row & (SEQ-1);
        const int head = col >> 8, sub = col & 255;      // kv: [head][2][128]
        const int gcol = (head >> 2)*GD + (head & 3)*HD; // G = head/4, hh = head%4
        if (sub < HD) Kb[(size_t)row*DIM + gcol + sub] = f32_to_bf16(val);
        else          Vt[((size_t)b*DIM + gcol + (sub - HD))*SEQ + lk] = f32_to_bf16(val);
      } else {
        outF[(size_t)row*N + col] = val;
      }
    }
  }
}

// ---------------------------------------------------------------- attention
// One block = 16 queries of (b,g). 4 waves; wave w owns head h=w (128 output
// cols, 8 f32 C tiles in VGPRs). The 16x512 Q tile is staged into LDS once
// per block by the Tensor Data Mover (s_wait_tensorcnt sync), then streamed
// against 4096 keys (G-major) in 64-key blocks with online softmax.
__global__ void attn_kernel(const u16* __restrict__ Qb, const u16* __restrict__ Kb,
                            const u16* __restrict__ Vt,
                            const unsigned char* __restrict__ mask,
                            u16* __restrict__ Ob){
  __shared__ u16   s_q[16*GD];          // 16 KB Q tile
  __shared__ float s_S[16*64];
  __shared__ float s_m[16], s_l[16], s_fac[16];
  const int tid = threadIdx.x, wv = tid >> 5, ln = tid & 31;
  const int lane16 = ln & 15, hi8 = (ln >= 16) ? 8 : 0;
  const int qt = blockIdx.x & 127;
  const int g  = (blockIdx.x >> 7) & 1;
  const int b  = blockIdx.x >> 8;
  const int q0 = qt*16;

  // ---- stage Q tile (rows q0..q0+15, cols g*512..+512) into LDS ----
#if __has_builtin(__builtin_amdgcn_tensor_load_to_lds)
  if (wv == 0){
    const unsigned long long ga =
        (unsigned long long)(uintptr_t)(Qb + (size_t)(b*SEQ + q0)*DIM + g*GD);
    const unsigned lds_off = (unsigned)(uintptr_t)(void*)s_q;  // LDS offset = addr[31:0]
    v4u g0;
    g0[0] = 1u;                                   // count=1, user descriptor
    g0[1] = lds_off;                              // lds_addr
    g0[2] = (unsigned)(ga & 0xFFFFFFFFu);         // global_addr[31:0]
    g0[3] = (unsigned)((ga >> 32) & 0x01FFFFFFu) | (2u << 30);  // addr[56:32], type=2
    v8i g1;
    g1[0] = (int)(1u << 16);                      // data_size=1 (2 bytes)
    g1[1] = (int)((1024u & 0xFFFFu) << 16);       // tensor_dim0[15:0]  (bits 63:48)
    g1[2] = (int)((1024u >> 16) | (16u << 16));   // tensor_dim0[31:16], tensor_dim1[15:0]
    g1[3] = (int)(512u << 16);                    // tensor_dim1[31:16]=0, tile_dim0=512
    g1[4] = (int)(16u);                           // tile_dim1=16, tile_dim2=0
    g1[5] = (int)(1024u);                         // tensor_dim0_stride[31:0]=1024
    g1[6] = 0;                                    // stride0[47:32], dim1_stride[15:0]
    g1[7] = 0;
    v4i z4; z4[0] = z4[1] = z4[2] = z4[3] = 0;
#if __clang_major__ >= 23
    v8i z8; 
#pragma unroll
    for (int i = 0; i < 8; ++i) z8[i] = 0;
    __builtin_amdgcn_tensor_load_to_lds(g0, g1, z4, z4, z8, 0);
#else
    __builtin_amdgcn_tensor_load_to_lds(g0, g1, z4, z4, 0);
#endif
    __builtin_amdgcn_s_wait_tensorcnt(0);
  }
#else
  // fallback: cooperative vector copy global -> LDS
  for (int i = tid; i < 16*GD/8; i += 128){
    const int row = i >> 6, c8 = i & 63;
    *(v8bf*)(s_q + row*GD + c8*8) =
        *(const v8bf*)(Qb + (size_t)(b*SEQ + q0 + row)*DIM + g*GD + c8*8);
  }
#endif
  if (tid < 16){ s_m[tid] = -3.0e38f; s_l[tid] = 0.f; }
  __syncthreads();

  // resident Q fragments from LDS (ds_load_b128), 16 chunks of K=32
  v16bf qf[16];
  {
    const u16* qlds = s_q + lane16*GD + hi8;
#pragma unroll
    for (int c = 0; c < 16; ++c) qf[c] = ld_frag(qlds + c*32);
  }
  v8f o[8];
#pragma unroll
  for (int t = 0; t < 8; ++t)
#pragma unroll
    for (int v = 0; v < 8; ++v) o[t][v] = 0.f;

  for (int kb = 0; kb < NKEY/64; ++kb){
    // ---- S: this wave's 16-key stripe (full 512-dim contraction) ----
    const int kidx = kb*64 + wv*16;
    const int G   = kidx >> 11;          // key group (fractal index)
    const int lk0 = kidx & (SEQ-1);
    const u16* kbase = Kb + (size_t)(b*SEQ + lk0 + lane16)*DIM + G*GD + hi8;
    if (kb + 1 < NKEY/64){               // L2 warm-up for next key block
      const int k2 = (kb+1)*64 + wv*16;
      __builtin_prefetch(Kb + (size_t)(b*SEQ + (k2 & (SEQ-1)) + lane16)*DIM
                            + (k2 >> 11)*GD, 0, 1);
    }
    v8f s;
#pragma unroll
    for (int v = 0; v < 8; ++v) s[v] = 0.f;
    {
      v16bf kf = ld_frag(kbase);         // software-pipelined K fragments
#pragma unroll
      for (int c = 0; c < 16; ++c){
        v16bf kn = kf;
        if (c < 15) kn = ld_frag(kbase + (c+1)*32);
        s = wmma_bf16(qf[c], kf, s);
        kf = kn;
      }
    }
    const float sc = SCALE * (G ? 2.0f : 1.0f);     // FRACTAL_SCALE^G
    const bool mk = mask[b*SEQ + lk0 + lane16] != 0;
#pragma unroll
    for (int v = 0; v < 8; ++v)
      s_S[(v + hi8)*64 + wv*16 + lane16] = mk ? -10000.f : s[v]*sc;
    __syncthreads();

    // ---- online softmax stats: one row per thread 0..15 ----
    if (tid < 16){
      const int r = tid;
      float m = s_m[r], mb = -3.0e38f;
      for (int j = 0; j < 64; ++j) mb = fmaxf(mb, s_S[r*64 + j]);
      const float mn = fmaxf(m, mb);
      const float fac = __expf(m - mn);
      float ls = 0.f;
      for (int j = 0; j < 64; ++j){
        float p = __expf(s_S[r*64 + j] - mn);
        s_S[r*64 + j] = p;
        ls += p;
      }
      s_l[r] = s_l[r]*fac + ls;
      s_m[r] = mn;
      s_fac[r] = fac;
    }
    __syncthreads();

    // ---- rescale accumulators ----
#pragma unroll
    for (int t = 0; t < 8; ++t)
#pragma unroll
      for (int v = 0; v < 8; ++v) o[t][v] *= s_fac[v + hi8];

    // ---- O += P(16x64) @ V(64x128-slice), two K=32 chunks ----
#pragma unroll
    for (int kc = 0; kc < 2; ++kc){
      v16bf pf;
      {
        const int kofs = kc*32 + hi8;
#pragma unroll
        for (int j = 0; j < 8; ++j){
          pf[j]   = us_as_bf(f32_to_bf16(s_S[lane16*64 + kofs + j]));
          pf[j+8] = us_as_bf(f32_to_bf16(s_S[lane16*64 + kofs + 16 + j]));
        }
      }
      const int kv  = kb*64 + kc*32;
      const int Gv  = kv >> 11;
      const int lv0 = kv & (SEQ-1);
      // Vt is [b][col][Lk] so B-fragments are contiguous along keys
      const u16* vcol = Vt + ((size_t)b*DIM + Gv*GD + wv*HD + lane16)*SEQ
                        + lv0 + hi8;
      v16bf vf = ld_frag(vcol);          // software-pipelined V fragments
#pragma unroll
      for (int t = 0; t < 8; ++t){
        v16bf vn = vf;
        if (t < 7) vn = ld_frag(vcol + (size_t)(t+1)*16*SEQ);
        o[t] = wmma_bf16(pf, vf, o[t]);
        vf = vn;
      }
    }
    __syncthreads();
  }

  // ---- normalize by row-sum, store bf16 (feeds output-projection GEMM) ----
#pragma unroll
  for (int t = 0; t < 8; ++t){
    const int col = g*GD + wv*HD + t*16 + lane16;
#pragma unroll
    for (int v = 0; v < 8; ++v){
      const int row = b*SEQ + q0 + v + hi8;
      Ob[(size_t)row*DIM + col] = f32_to_bf16(o[t][v] / s_l[v + hi8]);
    }
  }
}

// ---------------------------------------------------------------- LayerNorm
__global__ void ln_kernel(const float* __restrict__ x, const float* __restrict__ r,
                          const float* __restrict__ gma, const float* __restrict__ bta,
                          float* __restrict__ y){
  __shared__ float sh1[256], sh2[256];
  const int row = blockIdx.x, tid = threadIdx.x;
  const float* xr = x + (size_t)row*DIM;
  const float* rr = r + (size_t)row*DIM;
  float s = 0.f, s2 = 0.f;
  for (int i = tid; i < DIM; i += 256){
    float t = xr[i] + rr[i];
    s += t; s2 += t*t;
  }
  sh1[tid] = s; sh2[tid] = s2; __syncthreads();
  for (int off = 128; off > 0; off >>= 1){
    if (tid < off){ sh1[tid] += sh1[tid+off]; sh2[tid] += sh2[tid+off]; }
    __syncthreads();
  }
  const float mu  = sh1[0] * (1.0f/DIM);
  const float var = sh2[0] * (1.0f/DIM) - mu*mu;
  const float inv = rsqrtf(var + LN_EPS);
  for (int i = tid; i < DIM; i += 256){
    float t = xr[i] + rr[i];
    y[(size_t)row*DIM + i] = (t - mu)*inv*gma[i] + bta[i];
  }
}

// ---------------------------------------------------------------- launch
extern "C" void kernel_launch(void* const* d_in, const int* in_sizes, int n_in,
                              void* d_out, int out_size, void* d_ws, size_t ws_size,
                              hipStream_t stream){
  const float* x    = (const float*)d_in[0];
  const unsigned char* mask = (const unsigned char*)d_in[1]; // jax bool -> 1 byte
  const float* Wq   = (const float*)d_in[2];
  const float* Wkv  = (const float*)d_in[3];
  const float* Wo   = (const float*)d_in[4];
  const float* ln_g = (const float*)d_in[5];
  const float* ln_b = (const float*)d_in[6];
  float* out = (float*)d_out;
  (void)in_sizes; (void)n_in; (void)out_size; (void)ws_size;

  char* w = (char*)d_ws;
  size_t off = 0;
  auto take = [&](size_t bytes)->char*{
    char* p = w + off; off = (off + bytes + 255) & ~(size_t)255; return p;
  };
  const size_t MROWS = (size_t)BATCH*SEQ;                 // 4096
  u16* xb  = (u16*)take(MROWS*DIM*2);
  u16* wqb = (u16*)take((size_t)DIM*DIM*2);
  u16* wkb = (u16*)take((size_t)2*DIM*DIM*2);
  u16* wob = (u16*)take((size_t)DIM*DIM*2);
  u16* Qb  = (u16*)take(MROWS*DIM*2);
  u16* Kb  = (u16*)take(MROWS*DIM*2);
  u16* Vt  = (u16*)take((size_t)BATCH*DIM*SEQ*2);
  u16* Ob  = (u16*)take(MROWS*DIM*2);
  float* R = (float*)take(MROWS*DIM*4);

  cvt_f32_bf16_kernel<<<dim3(1024),256,0,stream>>>(x,   xb,  (int)(MROWS*DIM));
  cvt_f32_bf16_kernel<<<dim3(256), 256,0,stream>>>(Wq,  wqb, DIM*DIM);
  cvt_f32_bf16_kernel<<<dim3(512), 256,0,stream>>>(Wkv, wkb, 2*DIM*DIM);
  cvt_f32_bf16_kernel<<<dim3(256), 256,0,stream>>>(Wo,  wob, DIM*DIM);

  gemm_bf16_kernel<<<dim3(64,16),128,0,stream>>>(xb, wqb, (int)MROWS, DIM,   DIM, 0,
                                                 Qb, nullptr, nullptr, nullptr);
  gemm_bf16_kernel<<<dim3(64,32),128,0,stream>>>(xb, wkb, (int)MROWS, 2*DIM, DIM, 1,
                                                 nullptr, Kb, Vt, nullptr);
  attn_kernel<<<dim3(BATCH*QG*(SEQ/16)),128,0,stream>>>(Qb, Kb, Vt, mask, Ob);
  gemm_bf16_kernel<<<dim3(64,16),128,0,stream>>>(Ob, wob, (int)MROWS, DIM,   DIM, 2,
                                                 nullptr, nullptr, nullptr, R);
  ln_kernel<<<dim3((int)MROWS),256,0,stream>>>(x, R, ln_g, ln_b, out);
}
